// labelG_15942918603370
// MI455X (gfx1250) — matrix operand order repaired and checked by reading, not compiled, not run on hardware
//
#include <hip/hip_runtime.h>
#include <hip/hip_bf16.h>

typedef __attribute__((ext_vector_type(16))) __bf16 v16bf;
typedef __attribute__((ext_vector_type(8)))  float  v8f;

#define K_DIM 200
#define NSTEP 7      // ceil(200/32)
#define C_DIM 64
#define EPS_BN 1e-5f

// ---------------- init: deg = 1 (self loop), stats = 0 ----------------
__global__ void k_init(float* __restrict__ deg, float* __restrict__ stats, int N) {
    int g = blockIdx.x * blockDim.x + threadIdx.x;
    if (g < N) deg[g] = 1.0f;
    if (g < 128) stats[g] = 0.0f;
}

// ---------------- degree scatter: deg[dst] += 1 ----------------
__global__ void k_deg(const long long* __restrict__ dst, float* __restrict__ deg, long long E) {
    long long e = (long long)blockIdx.x * blockDim.x + threadIdx.x;
    if (e < E) atomicAdd(&deg[dst[e]], 1.0f);
}

// ---------------- dinv = rsqrt(deg) in place ----------------
__global__ void k_rsqrt(float* __restrict__ d, int N) {
    int g = blockIdx.x * blockDim.x + threadIdx.x;
    if (g < N) d[g] = rsqrtf(d[g]);
}

// ---------------- WMMA GEMM helpers ----------------
__device__ __forceinline__ void cvt_hilo(const float a[16], v16bf& hi, v16bf& lo) {
#pragma unroll
    for (int i = 0; i < 16; ++i) {
        __bf16 h = (__bf16)a[i];
        hi[i] = h;
        lo[i] = (__bf16)(a[i] - (float)h);
    }
}

__device__ __forceinline__ void wmma_step(int ks, int lane, const v16bf& ahi, const v16bf& alo,
                                          const __bf16* __restrict__ sBhi,
                                          const __bf16* __restrict__ sBlo, v8f acc[4]) {
#pragma unroll
    for (int t = 0; t < 4; ++t) {
        const int off = (((ks * 4 + t) * 32) + lane) * 16;   // 32B-aligned contiguous fragment
        v16bf bhi = *(const v16bf*)(sBhi + off);
        v16bf blo = *(const v16bf*)(sBlo + off);
        acc[t] = __builtin_amdgcn_wmma_f32_16x16x32_bf16(false, ahi, false, bhi,
                                                         (short)0, acc[t], false, false);
        acc[t] = __builtin_amdgcn_wmma_f32_16x16x32_bf16(false, ahi, false, blo,
                                                         (short)0, acc[t], false, false);
        acc[t] = __builtin_amdgcn_wmma_f32_16x16x32_bf16(false, alo, false, bhi,
                                                         (short)0, acc[t], false, false);
    }
}

// ---------------- WMMA GEMM: xw = x @ W  (bf16 hi/lo split, f32 accum) ----------------
// W is pre-swizzled in LDS into per-lane B-fragment layout:
//   sB[ks][t][lane][i]  holds  W[ks*32 + (lane>>4)*8 + (i&7) + (i>>3)*16][t*16 + (lane&15)]
// so each fragment is one contiguous, 32B-aligned 16 x bf16 run per lane.
__global__ __launch_bounds__(256) void k_gemm_wmma(const float* __restrict__ x,
                                                   const float* __restrict__ W,
                                                   float* __restrict__ xw, int N) {
    __shared__ __attribute__((aligned(32))) __bf16 sBhi[NSTEP * 4 * 32 * 16];  // 28672 B
    __shared__ __attribute__((aligned(32))) __bf16 sBlo[NSTEP * 4 * 32 * 16];  // 28672 B
    const int tid = threadIdx.x;

    for (int idx = tid; idx < NSTEP * 4 * 32 * 16; idx += 256) {
        int i    = idx & 15;
        int lane = (idx >> 4) & 31;
        int t    = (idx >> 9) & 3;
        int ks   = idx >> 11;
        int k    = ks * 32 + (lane >> 4) * 8 + (i & 7) + (i >> 3) * 16;
        int c    = t * 16 + (lane & 15);
        float w  = (k < K_DIM) ? W[k * C_DIM + c] : 0.0f;
        __bf16 h = (__bf16)w;
        sBhi[idx] = h;
        sBlo[idx] = (__bf16)(w - (float)h);
    }
    __syncthreads();

    const int wave  = tid >> 5;          // wave32
    const int lane  = tid & 31;
    const int l16   = lane & 15;
    const int hiGrp = lane >> 4;         // 0 | 1
    const int m0    = blockIdx.x * 128 + wave * 16;
    const int row   = m0 + l16;
    const int rowc  = row < N ? row : (N - 1);
    const float* xr = x + (size_t)rowc * K_DIM;

    v8f acc[4] = {v8f{}, v8f{}, v8f{}, v8f{}};

    // full k-steps: unconditional vector loads
    for (int ks = 0; ks < 6; ++ks) {
        const int kbase = ks * 32 + hiGrp * 8;
        float4 q0 = *(const float4*)(xr + kbase);
        float4 q1 = *(const float4*)(xr + kbase + 4);
        float4 q2 = *(const float4*)(xr + kbase + 16);
        float4 q3 = *(const float4*)(xr + kbase + 20);
        float a[16] = {q0.x, q0.y, q0.z, q0.w, q1.x, q1.y, q1.z, q1.w,
                       q2.x, q2.y, q2.z, q2.w, q3.x, q3.y, q3.z, q3.w};
        v16bf ahi, alo;
        cvt_hilo(a, ahi, alo);
        wmma_step(ks, lane, ahi, alo, sBhi, sBlo, acc);
    }
    // epilogue k-step (K = 192..223; only K=192..199 valid, lanes 0-15 chunk 0)
    {
        float a[16];
#pragma unroll
        for (int i = 0; i < 16; ++i) a[i] = 0.0f;
        if (hiGrp == 0) {
            float4 q0 = *(const float4*)(xr + 192);
            float4 q1 = *(const float4*)(xr + 196);
            a[0] = q0.x; a[1] = q0.y; a[2] = q0.z; a[3] = q0.w;
            a[4] = q1.x; a[5] = q1.y; a[6] = q1.z; a[7] = q1.w;
        }
        v16bf ahi, alo;
        cvt_hilo(a, ahi, alo);
        wmma_step(6, lane, ahi, alo, sBhi, sBlo, acc);
    }

    // C/D layout: VGPR j -> row m0 + j + 8*hiGrp, col = t*16 + l16
    if (m0 + 16 <= N) {
#pragma unroll
        for (int t = 0; t < 4; ++t)
#pragma unroll
            for (int j = 0; j < 8; ++j)
                xw[(size_t)(m0 + j + hiGrp * 8) * C_DIM + t * 16 + l16] = acc[t][j];
    } else {
#pragma unroll
        for (int t = 0; t < 4; ++t)
#pragma unroll
            for (int j = 0; j < 8; ++j) {
                int m = m0 + j + hiGrp * 8;
                if (m < N) xw[(size_t)m * C_DIM + t * 16 + l16] = acc[t][j];
            }
    }
}

// ---------------- h = b + dinv^2 * xw (bias + self-loop message) ----------------
__global__ void k_init_h(const float* __restrict__ xw, const float* __restrict__ dinv,
                         const float* __restrict__ b, float* __restrict__ h, int N) {
    size_t g = (size_t)blockIdx.x * blockDim.x + threadIdx.x;
    if (g >= (size_t)N * C_DIM) return;
    int i = (int)(g >> 6), c = (int)(g & 63);
    float di = dinv[i];
    h[g] = b[c] + di * di * xw[g];
}

// ---------------- edge scatter: h[dst] += dinv[src]*dinv[dst] * xw[src] ----------------
// 16 threads per edge, 4 channels each (float4 gather + 4 f32 atomics; h lives in L2)
__global__ void k_scatter(const long long* __restrict__ src, const long long* __restrict__ dst,
                          const float* __restrict__ dinv, const float* __restrict__ xw,
                          float* __restrict__ h, long long E) {
    long long g = (long long)blockIdx.x * blockDim.x + threadIdx.x;
    long long e = g >> 4;
    if (e >= E) return;
    int c4 = (int)(g & 15) * 4;
    long long s = src[e], d = dst[e];
    float nrm = dinv[s] * dinv[d];
    const float4 v = *(const float4*)(xw + (size_t)s * C_DIM + c4);
    float* hp = h + (size_t)d * C_DIM + c4;
    atomicAdd(hp + 0, nrm * v.x);
    atomicAdd(hp + 1, nrm * v.y);
    atomicAdd(hp + 2, nrm * v.z);
    atomicAdd(hp + 3, nrm * v.w);
}

// ---------------- BN stats: stats[c] = sum relu(h), stats[64+c] = sum relu(h)^2 ----------------
__global__ __launch_bounds__(256) void k_bn_stats(const float* __restrict__ h,
                                                  float* __restrict__ stats, int N) {
    int c = threadIdx.x & 63;
    int grp = threadIdx.x >> 6;    // 0..3
    float s = 0.0f, s2 = 0.0f;
    for (int i = blockIdx.x * 4 + grp; i < N; i += gridDim.x * 4) {
        float v = fmaxf(h[(size_t)i * C_DIM + c], 0.0f);
        s += v; s2 += v * v;
    }
    __shared__ float red[2][4][64];
    red[0][grp][c] = s;
    red[1][grp][c] = s2;
    __syncthreads();
    if (grp == 0) {
        s  = red[0][0][c] + red[0][1][c] + red[0][2][c] + red[0][3][c];
        s2 = red[1][0][c] + red[1][1][c] + red[1][2][c] + red[1][3][c];
        atomicAdd(&stats[c], s);
        atomicAdd(&stats[64 + c], s2);
    }
}

// ---------------- BN apply (in place on d_out): y = (relu(h)-mean)*rsqrt(var+eps)*w + b ----------------
__global__ void k_bn_apply(float* __restrict__ h, const float* __restrict__ stats,
                           const float* __restrict__ bnw, const float* __restrict__ bnb, int N) {
    size_t g = (size_t)blockIdx.x * blockDim.x + threadIdx.x;
    if (g >= (size_t)N * C_DIM) return;
    int c = (int)(g & 63);
    float invN = 1.0f / (float)N;
    float mean = stats[c] * invN;
    float var  = stats[64 + c] * invN - mean * mean;
    float v = fmaxf(h[g], 0.0f);
    h[g] = (v - mean) * rsqrtf(var + EPS_BN) * bnw[c] + bnb[c];
}

extern "C" void kernel_launch(void* const* d_in, const int* in_sizes, int n_in,
                              void* d_out, int out_size, void* d_ws, size_t ws_size,
                              hipStream_t stream) {
    const float*     x   = (const float*)d_in[0];
    const long long* ei  = (const long long*)d_in[1];   // int64 [2, E]
    const float*     W   = (const float*)d_in[2];
    const float*     b   = (const float*)d_in[3];
    const float*     bnw = (const float*)d_in[4];
    const float*     bnb = (const float*)d_in[5];

    const int       N = in_sizes[0] / K_DIM;
    const long long E = (long long)in_sizes[1] / 2;
    const long long* srcp = ei;
    const long long* dstp = ei + E;

    // workspace layout: xw [N*64] | dinv [N] | stats [128]
    float* xw    = (float*)d_ws;
    float* dinv  = xw + (size_t)N * C_DIM;
    float* stats = dinv + N;
    float* h     = (float*)d_out;   // accumulate h in d_out, BN applied in place

    const int T = 256;
    int gN  = (N + T - 1) / T;
    int gE  = (int)((E + T - 1) / T);
    long long totNC = (long long)N * C_DIM;
    int gNC = (int)((totNC + T - 1) / T);
    long long totSc = E * 16;
    int gSc = (int)((totSc + T - 1) / T);

    k_init      <<<gN,  T, 0, stream>>>(dinv, stats, N);
    k_deg       <<<gE,  T, 0, stream>>>(dstp, dinv, E);
    k_rsqrt     <<<gN,  T, 0, stream>>>(dinv, N);
    k_gemm_wmma <<<(N + 127) / 128, T, 0, stream>>>(x, W, xw, N);
    k_init_h    <<<gNC, T, 0, stream>>>(xw, dinv, b, h, N);
    k_scatter   <<<gSc, T, 0, stream>>>(srcp, dstp, dinv, xw, h, E);
    k_bn_stats  <<<512, T, 0, stream>>>(h, stats, N);
    k_bn_apply  <<<gNC, T, 0, stream>>>(h, stats, bnw, bnb, N);
}
